// TangoMamba_27685359190705
// MI455X (gfx1250) — matrix operand order repaired and checked by reading, not compiled, not run on hardware
//
#include <hip/hip_runtime.h>
#include <hip/hip_bf16.h>
#include <math.h>

typedef _Float16 v16h __attribute__((ext_vector_type(16)));
typedef _Float16 v8h  __attribute__((ext_vector_type(8)));
typedef float    v8f  __attribute__((ext_vector_type(8)));
typedef float    v4f  __attribute__((ext_vector_type(4)));

namespace {
constexpr int NB    = 4;      // batch
constexpr int LL    = 1024;   // sequence length
constexpr int DM    = 512;    // d_model
constexpr int DS    = 16;     // d_state
constexpr int DCONV = 4;
constexpr int DI    = 1024;   // d_inner
constexpr int DTR   = 32;     // dt_rank
constexpr int NL    = 2;
constexpr int BL    = NB * LL;   // 4096 GEMM rows

constexpr int BM = 128, BN = 64, BK = 32;   // block tile, 8 waves (256 thr)

enum { EPI_NONE = 0, EPI_SOFTPLUS_BIAS = 1, EPI_RESID = 2 };
}

__device__ __forceinline__ float fast_sigmoid(float x) {
  return 1.0f / (1.0f + __expf(-x));
}

// ---------------------------------------------------------------------------
// WMMA GEMM: C[M,N] = A[M,K] * W[N,K]^T  (f32 in, f16 WMMA, f32 accumulate)
// Block tile 128x64, K-step 32, 8 waves; double-buffered LDS ping-pong with a
// single barrier per K-step; next tile's global loads issued before the WMMAs.
// ---------------------------------------------------------------------------
template <int EPI>
__global__ __launch_bounds__(256) void wmma_gemm_kernel(
    const float* __restrict__ A, int lda,
    const float* __restrict__ W, int ldw,
    float* __restrict__ C, int ldc, int K,
    const float* __restrict__ bias,
    const float* __restrict__ resid)
{
  __shared__ __align__(32) _Float16 sA[2][BM * BK];   // 2 x 8 KB
  __shared__ __align__(32) _Float16 sW[2][BN * BK];   // 2 x 4 KB

  const int tid  = threadIdx.x;
  const int lane = tid & 31;
  const int wave = tid >> 5;
  const int m0   = blockIdx.y * BM;
  const int n0   = blockIdx.x * BN;
  const int wrow = wave * 16;            // wave's 16-row slice in block tile

  v8f acc[4] = {};                       // 4 column subtiles of 16x16

  v4f aReg[4];                           // staging: 128x32 A tile (1024 f4)
  v4f wReg[2];                           // staging:  64x32 W tile ( 512 f4)

  auto load_tiles = [&](int kk) {
    #pragma unroll
    for (int i = 0; i < 4; ++i) {
      int idx = tid + i * 256;
      int r = idx >> 3, c = (idx & 7) << 2;
      aReg[i] = *(const v4f*)(A + (size_t)(m0 + r) * lda + kk + c);
    }
    #pragma unroll
    for (int i = 0; i < 2; ++i) {
      int idx = tid + i * 256;
      int r = idx >> 3, c = (idx & 7) << 2;
      wReg[i] = *(const v4f*)(W + (size_t)(n0 + r) * ldw + kk + c);
    }
  };
  auto stage = [&](int p) {
    #pragma unroll
    for (int i = 0; i < 4; ++i) {
      int idx = tid + i * 256;
      int r = idx >> 3, c = (idx & 7) << 2;
      #pragma unroll
      for (int e = 0; e < 4; ++e) sA[p][r * BK + c + e] = (_Float16)aReg[i][e];
    }
    #pragma unroll
    for (int i = 0; i < 2; ++i) {
      int idx = tid + i * 256;
      int r = idx >> 3, c = (idx & 7) << 2;
      #pragma unroll
      for (int e = 0; e < 4; ++e) sW[p][r * BK + c + e] = (_Float16)wReg[i][e];
    }
  };

  // Prologue: stage first K tile into buffer 0.
  load_tiles(0);
  stage(0);

  // Fragment indices (A 16x32 / B 32x16, both in contiguous k = 16*(lane>>4) + s order).
  const int arow = wrow + (lane & 15);
  const int ksel = lane >> 4;
  const int bcol = lane & 15;
  const int kb   = (lane >> 4) * 16;

  int p = 0;
  for (int k0 = 0; k0 < K; k0 += BK) {
    __syncthreads();                     // buf p ready; buf p^1 free
    const bool hasNext = (k0 + BK) < K;
    if (hasNext) load_tiles(k0 + BK);    // issue global loads early
    if (k0 + 2 * BK < K) {               // warm L2/L0 two tiles ahead
      int pr = tid & 63;
      __builtin_prefetch(A + (size_t)(m0 + pr) * lda + (k0 + 2 * BK), 0, 0);
      __builtin_prefetch(W + (size_t)(n0 + pr) * ldw + (k0 + 2 * BK), 0, 0);
    }

    union { v16h v; v8h h[2]; } af;
    af.h[0] = *(const v8h*)(&sA[p][arow * BK + ksel * 16]);
    af.h[1] = *(const v8h*)(&sA[p][arow * BK + ksel * 16 + 8]);
    #pragma unroll
    for (int j = 0; j < 4; ++j) {
      v16h bf = *(const v16h*)(&sW[p][(j * 16 + bcol) * BK + kb]);
      acc[j] = __builtin_amdgcn_wmma_f32_16x16x32_f16(
          false, af.v, false, bf, (short)0, acc[j], false, false);
    }

    if (hasNext) stage(p ^ 1);           // convert + store behind the WMMAs
    p ^= 1;
  }

  // Epilogue + writeback. C/D layout: lane = N + 16*(M>=8), VGPR r = M%8.
  const int n     = lane & 15;
  const int mhalf = (lane >> 4) * 8;
  #pragma unroll
  for (int j = 0; j < 4; ++j) {
    #pragma unroll
    for (int r = 0; r < 8; ++r) {
      int gm = m0 + wrow + mhalf + r;
      int gn = n0 + j * 16 + n;
      float v = acc[j][r];
      if (EPI == EPI_SOFTPLUS_BIAS) {
        v += bias[gn];
        v = (v > 20.0f) ? v : log1pf(__expf(v));
      }
      if (EPI == EPI_RESID) {
        v += resid[(size_t)gm * ldc + gn];
      }
      C[(size_t)gm * ldc + gn] = v;
    }
  }
}

// ---------------------------------------------------------------------------
// Depthwise causal conv (width 4) + SiLU.
// ---------------------------------------------------------------------------
__global__ __launch_bounds__(256) void conv_silu_kernel(
    const float* __restrict__ xz, const float* __restrict__ cw,
    const float* __restrict__ cb, float* __restrict__ xh)
{
  int idx = blockIdx.x * 256 + threadIdx.x;   // over NB*LL*DI
  int d = idx % DI;
  int t = (idx / DI) % LL;
  int b = idx / (DI * LL);
  float acc = cb[d];
  #pragma unroll
  for (int k = 0; k < DCONV; ++k) {
    int ti = t + k - (DCONV - 1);
    if (ti >= 0)
      acc += cw[d * DCONV + k] * xz[(size_t)(b * LL + ti) * (2 * DI) + d];
  }
  xh[idx] = acc * fast_sigmoid(acc);
}

// ---------------------------------------------------------------------------
// Selective scan. One thread owns one channel: h[16], A[16] in registers.
// B_t/C_t (32 floats/step) are double-buffered in LDS and fetched one step
// ahead with CDNA5 async global->LDS copies (ASYNCcnt), overlapping the
// broadcast load with the recurrence math.
// ---------------------------------------------------------------------------
__global__ __launch_bounds__(256) void scan_kernel(
    const float* __restrict__ xh,
    const float* __restrict__ xz,      // z lives at column DI + d
    const float* __restrict__ xdbl,    // row stride 64; B at 32..47, C at 48..63
    const float* __restrict__ dtb,
    const float* __restrict__ A_log,
    const float* __restrict__ Dp,
    float* __restrict__ y)
{
  const int tid = threadIdx.x;
  const int d = blockIdx.x * blockDim.x + tid;           // 0..DI-1
  const int b = blockIdx.y;
  __shared__ float sBC[2][2 * DS];

  float h[DS];
  float An[DS];
  #pragma unroll
  for (int n = 0; n < DS; ++n) {
    h[n]  = 0.0f;
    An[n] = -__expf(A_log[d * DS + n]);
  }
  const float dval = Dp[d];

  const float* dt_row = dtb  + (size_t)b * LL * DI + d;
  const float* xh_row = xh   + (size_t)b * LL * DI + d;
  const float* z_row  = xz   + (size_t)b * LL * (2 * DI) + DI + d;
  const float* bc_row = xdbl + (size_t)b * LL * 64 + DTR;
  float*       y_row  = y    + (size_t)b * LL * DI + d;

  // Preload t=0 (wave 0, one async b32 per lane; LDS addr = low 32 bits).
  if (tid < 2 * DS) {
    unsigned int l0 = (unsigned int)(uintptr_t)&sBC[0][tid];
    unsigned long long g0 = (unsigned long long)(uintptr_t)(bc_row + tid);
    asm volatile("global_load_async_to_lds_b32 %0, %1, off"
                 :: "v"(l0), "v"(g0) : "memory");
  }

  int p = 0;
  for (int t = 0; t < LL; ++t) {
    if (tid < 2 * DS)
      asm volatile("s_wait_asynccnt 0x0" ::: "memory");  // wave0: buf p landed
    __syncthreads();                                     // publish buf p

    // Prefetch next step's B/C into the alternate buffer (no reader conflict:
    // buf p^1's last readers passed the barrier above).
    if (t + 1 < LL && tid < 2 * DS) {
      unsigned int l1 = (unsigned int)(uintptr_t)&sBC[p ^ 1][tid];
      unsigned long long g1 =
          (unsigned long long)(uintptr_t)(bc_row + (size_t)(t + 1) * 64 + tid);
      asm volatile("global_load_async_to_lds_b32 %0, %1, off"
                   :: "v"(l1), "v"(g1) : "memory");
    }

    float dt_v = dt_row[(size_t)t * DI];
    float x_v  = xh_row[(size_t)t * DI];
    float z_v  = z_row[(size_t)t * (2 * DI)];
    float dtx  = dt_v * x_v;
    float accv = 0.0f;
    #pragma unroll
    for (int n = 0; n < DS; ++n) {
      float dA = __expf(dt_v * An[n]);
      h[n] = dA * h[n] + dtx * sBC[p][n];
      accv += h[n] * sBC[p][DS + n];
    }
    accv = (accv + dval * x_v) * (z_v * fast_sigmoid(z_v));
    y_row[(size_t)t * DI] = accv;
    p ^= 1;
  }
}

// ---------------------------------------------------------------------------
// cur = x (dir 0) or time-flipped x (dir 1)
// ---------------------------------------------------------------------------
__global__ __launch_bounds__(256) void prep_kernel(
    const float* __restrict__ x, float* __restrict__ cur, int flip)
{
  int idx = blockIdx.x * 256 + threadIdx.x;   // over NB*LL*DM
  int d = idx % DM;
  int t = (idx / DM) % LL;
  int b = idx / (DM * LL);
  int ts = flip ? (LL - 1 - t) : t;
  cur[idx] = x[(size_t)(b * LL + ts) * DM + d];
}

// out = cur (first) or out += flip(cur)
__global__ __launch_bounds__(256) void writeback_kernel(
    const float* __restrict__ cur, float* __restrict__ out, int flip, int first)
{
  int idx = blockIdx.x * 256 + threadIdx.x;
  int d = idx % DM;
  int t = (idx / DM) % LL;
  int b = idx / (DM * LL);
  int ts = flip ? (LL - 1 - t) : t;
  float v = cur[(size_t)(b * LL + ts) * DM + d];
  if (first) out[idx] = v;
  else       out[idx] += v;
}

// ---------------------------------------------------------------------------
extern "C" void kernel_launch(void* const* d_in, const int* in_sizes, int n_in,
                              void* d_out, int out_size, void* d_ws, size_t ws_size,
                              hipStream_t stream) {
  (void)in_sizes; (void)n_in; (void)out_size; (void)ws_size;

  const float* x      = (const float*)d_in[0];
  const float* in_w   = (const float*)d_in[1];   // (NL, 2*DI, DM)
  const float* conv_w = (const float*)d_in[2];   // (NL, DI, DCONV)
  const float* conv_b = (const float*)d_in[3];   // (NL, DI)
  const float* xp_w   = (const float*)d_in[4];   // (NL, DTR+2*DS, DI)
  const float* dtp_w  = (const float*)d_in[5];   // (NL, DI, DTR)
  const float* dtp_b  = (const float*)d_in[6];   // (NL, DI)
  const float* A_log  = (const float*)d_in[7];   // (NL, DI, DS)
  const float* D_par  = (const float*)d_in[8];   // (NL, DI)
  const float* out_w  = (const float*)d_in[9];   // (NL, DM, DI)
  float* out = (float*)d_out;

  // Workspace carve-up (floats)
  float* ws   = (float*)d_ws;
  float* cur  = ws;                    size_t o = (size_t)BL * DM;
  float* xz   = ws + o;                o += (size_t)BL * 2 * DI;
  float* xh   = ws + o;                o += (size_t)BL * DI;
  float* xdbl = ws + o;                o += (size_t)BL * 64;
  float* dtb  = ws + o;                o += (size_t)BL * DI;
  float* ybuf = ws + o;                o += (size_t)BL * DI;

  const int ELEM_DI = (NB * LL * DI) / 256;
  const int ELEM_DM = (NB * LL * DM) / 256;

  for (int dir = 0; dir < 2; ++dir) {
    prep_kernel<<<ELEM_DM, 256, 0, stream>>>(x, cur, dir);

    for (int l = 0; l < NL; ++l) {
      const float* w_in  = in_w   + (size_t)l * 2 * DI * DM;
      const float* cw    = conv_w + (size_t)l * DI * DCONV;
      const float* cb    = conv_b + (size_t)l * DI;
      const float* w_xp  = xp_w   + (size_t)l * (DTR + 2 * DS) * DI;
      const float* w_dt  = dtp_w  + (size_t)l * DI * DTR;
      const float* b_dt  = dtp_b  + (size_t)l * DI;
      const float* al    = A_log  + (size_t)l * DI * DS;
      const float* dp    = D_par  + (size_t)l * DI;
      const float* w_out = out_w  + (size_t)l * DM * DI;

      // 1) xz = cur @ in_w^T            [4096 x 2048, K=512]
      wmma_gemm_kernel<EPI_NONE><<<dim3((2 * DI) / BN, BL / BM), 256, 0, stream>>>(
          cur, DM, w_in, DM, xz, 2 * DI, DM, nullptr, nullptr);

      // 2) xh = silu(conv(xz[:, :DI]))
      conv_silu_kernel<<<ELEM_DI, 256, 0, stream>>>(xz, cw, cb, xh);

      // 3) x_dbl = xh @ xp_w^T          [4096 x 64, K=1024]
      wmma_gemm_kernel<EPI_NONE><<<dim3(64 / BN, BL / BM), 256, 0, stream>>>(
          xh, DI, w_xp, DI, xdbl, 64, DI, nullptr, nullptr);

      // 4) dt = softplus(x_dbl[:, :32] @ dtp_w^T + b)   [4096 x 1024, K=32]
      wmma_gemm_kernel<EPI_SOFTPLUS_BIAS><<<dim3(DI / BN, BL / BM), 256, 0, stream>>>(
          xdbl, 64, w_dt, DTR, dtb, DI, DTR, b_dt, nullptr);

      // 5) selective scan -> y (includes +D*x and *silu(z))
      scan_kernel<<<dim3(DI / 256, NB), 256, 0, stream>>>(
          xh, xz, xdbl, dtb, al, dp, ybuf);

      // 6) cur = y @ out_w^T + cur      [4096 x 512, K=1024] (residual)
      wmma_gemm_kernel<EPI_RESID><<<dim3(DM / BN, BL / BM), 256, 0, stream>>>(
          ybuf, DI, w_out, DI, cur, DM, DI, nullptr, cur);
    }

    writeback_kernel<<<ELEM_DM, 256, 0, stream>>>(cur, out, dir, dir == 0 ? 1 : 0);
  }
}
